// MultiHead_88124138979764
// MI455X (gfx1250) — compile-verified, matrix operand order checked by reading, not run on hardware
//
#include <hip/hip_runtime.h>

// ---------------------------------------------------------------------------
// Multi-head attention forward for MI455X (gfx1250, wave32, WMMA bf16).
// B=4, S=2048, D=1024, H=16, DH=64.
// GEMM loops are 2-deep software-pipelined: fragments for step k+32 are loaded
// (one clause of global_load_b128) while the 4 WMMAs of step k execute.
// ---------------------------------------------------------------------------

#define B_ 4
#define S_ 2048
#define D_ 1024
#define H_ 16
#define DH_ 64

typedef __bf16 bf16;
typedef __attribute__((ext_vector_type(16))) __bf16 v16bf;
typedef __attribute__((ext_vector_type(8)))  float  v8f;

__device__ __forceinline__ bf16 f2bf(float f) {
    unsigned u = __builtin_bit_cast(unsigned, f);
    unsigned r = (u + 0x7FFFu + ((u >> 16) & 1u)) >> 16;  // round-to-nearest-even
    unsigned short s = (unsigned short)r;
    return __builtin_bit_cast(bf16, s);
}

__device__ __forceinline__ v16bf cvt16(const float* t) {
    v16bf r;
#pragma unroll
    for (int i = 0; i < 16; ++i) r[i] = f2bf(t[i]);
    return r;
}

__device__ __forceinline__ v16bf ldbf(const bf16* p) { return *(const v16bf*)p; }

__device__ __forceinline__ v8f vzero8() {
    v8f z = {0.f, 0.f, 0.f, 0.f, 0.f, 0.f, 0.f, 0.f};
    return z;
}

#define WMMA_BF16(A, Bm, C) \
    __builtin_amdgcn_wmma_f32_16x16x32_bf16(false, (A), false, (Bm), (short)0, (C), false, false)

// ---------------------------------------------------------------------------
// Kernel 1: convert weights to bf16.  Wq/Wk/Wv [H,D,64] -> transposed [H,64,D];
// Wo [D,D] kept row-major (its rows are the GEMM-B columns for out = X*Wo^T).
// ---------------------------------------------------------------------------
__global__ __launch_bounds__(256) void cvt_weights_kernel(
    const float* __restrict__ Wq, const float* __restrict__ Wk,
    const float* __restrict__ Wv, const float* __restrict__ Wo,
    bf16* __restrict__ wqt, bf16* __restrict__ wkt,
    bf16* __restrict__ wvt, bf16* __restrict__ wob) {
    const size_t WSZ = (size_t)H_ * D_ * DH_;          // 1,048,576
    const size_t TOT = 3 * WSZ + (size_t)D_ * D_;      // + Wo
    size_t stride = (size_t)gridDim.x * blockDim.x;
    for (size_t i = (size_t)blockIdx.x * blockDim.x + threadIdx.x; i < TOT; i += stride) {
        if (i < 3 * WSZ) {
            int m = (int)(i / WSZ);
            size_t r = i % WSZ;
            int h = (int)(r >> 16);            // r / (D_*DH_)
            int d = (int)((r >> 6) & (D_ - 1));
            int n = (int)(r & (DH_ - 1));
            const float* src = (m == 0) ? Wq : (m == 1) ? Wk : Wv;
            bf16* dst = (m == 0) ? wqt : (m == 1) ? wkt : wvt;
            dst[((size_t)h * DH_ + n) * D_ + d] = f2bf(src[r]);
        } else {
            size_t r = i - 3 * WSZ;
            wob[r] = f2bf(Wo[r]);
        }
    }
}

// ---------------------------------------------------------------------------
// Kernel 2: per-head projections.  For mat in {Q,K,V}:
//   C[128x64] = X[b, s0:s0+128, :] (f32->bf16 on the fly) * W_t[h].
// Q,K stored [bh, s, 64]; V stored transposed [bh, dv, S] for the P*V GEMM.
// grid = (S/128, B*H, 3), block = 256 (8 waves, 16 rows x 64 cols each).
// ---------------------------------------------------------------------------
__global__ __launch_bounds__(256) void proj_kernel(
    const float* __restrict__ q, const float* __restrict__ k,
    const float* __restrict__ v,
    const bf16* __restrict__ wqt, const bf16* __restrict__ wkt,
    const bf16* __restrict__ wvt,
    bf16* __restrict__ Qp, bf16* __restrict__ Kp, bf16* __restrict__ Vt) {
    const int mat = blockIdx.z;
    const int bh  = blockIdx.y;
    const int st  = blockIdx.x;
    const int b = bh >> 4, h = bh & 15;

    const float* src = (mat == 0) ? q : (mat == 1) ? k : v;
    const bf16*  wt  = (mat == 0) ? wqt : (mat == 1) ? wkt : wvt;

    const int w     = threadIdx.x >> 5;
    const int lane  = threadIdx.x & 31;
    const int lr    = lane & 15;
    const int kh    = (lane >> 4) << 4;   // k-half offset inside a 32-wide chunk
    const int rhalf = (lane >> 4) << 3;   // row-half offset for C tiles

    const int m0 = st * 128 + w * 16;
    const float* arow = src + ((size_t)b * S_ + m0 + lr) * D_;
    const bf16*  wrow = wt + ((size_t)h * DH_ + lr) * D_ + kh;  // + nt*16*D_ + kb

    v8f acc[4];
#pragma unroll
    for (int i = 0; i < 4; ++i) acc[i] = vzero8();

    auto ldA = [&](int kb, v16bf& af) {
        float tmp[16];
        const float4* ap = (const float4*)(arow + kb + kh);
#pragma unroll
        for (int i = 0; i < 4; ++i) *(float4*)(tmp + 4 * i) = ap[i];
        af = cvt16(tmp);
    };
    auto ldB = [&](int kb, v16bf bf[4]) {
#pragma unroll
        for (int nt = 0; nt < 4; ++nt)
            bf[nt] = ldbf(wrow + (size_t)nt * 16 * D_ + kb);
    };

    v16bf af0, af1, b0[4], b1[4];
    ldA(0, af0);
    ldB(0, b0);
#pragma unroll 1
    for (int kb = 0; kb < D_; kb += 64) {
        // stage k+32 while computing k
        ldA(kb + 32, af1);
        ldB(kb + 32, b1);
        if (kb + 96 < D_) __builtin_prefetch(arow + kb + 96, 0, 1);
#pragma unroll
        for (int nt = 0; nt < 4; ++nt) acc[nt] = WMMA_BF16(af0, b0[nt], acc[nt]);
        // stage k+64 while computing k+32
        if (kb + 64 < D_) {
            ldA(kb + 64, af0);
            ldB(kb + 64, b0);
        }
#pragma unroll
        for (int nt = 0; nt < 4; ++nt) acc[nt] = WMMA_BF16(af1, b1[nt], acc[nt]);
    }

    // Store C tiles (C layout: vgpr r, lanes 0-15 -> M=r, lanes 16-31 -> M=8+r).
    bf16* dstQK = (mat == 0) ? Qp : Kp;
#pragma unroll
    for (int nt = 0; nt < 4; ++nt) {
        const int col = nt * 16 + lr;
#pragma unroll
        for (int r = 0; r < 8; ++r) {
            const int row = m0 + r + rhalf;
            bf16 val = f2bf(acc[nt][r]);
            if (mat < 2) dstQK[((size_t)bh * S_ + row) * DH_ + col] = val;
            else         Vt[((size_t)bh * DH_ + col) * S_ + row]    = val;
        }
    }
}

// ---------------------------------------------------------------------------
// Kernel 3: causal flash attention per (b,h).  Each wave owns 16 query rows,
// streams key blocks of 32, online softmax, P staged via wave-private LDS to
// reshape the C-layout scores into an A-fragment for the P*V WMMAs.
// V fragments are loaded BEFORE the softmax so the ~100 VALU ops of
// mask/exp/shuffle reductions hide their latency.
// grid = (S/128, B*H), block = 256.
// ---------------------------------------------------------------------------
__global__ __launch_bounds__(256) void attn_kernel(
    const bf16* __restrict__ Qp, const bf16* __restrict__ Kp,
    const bf16* __restrict__ Vt, bf16* __restrict__ Xo) {
    __shared__ bf16 plds[8 * 16 * 32];   // 8 KB, one 16x32 P tile per wave

    const int bh = blockIdx.y;
    const int qt = blockIdx.x;
    const int w     = threadIdx.x >> 5;
    const int lane  = threadIdx.x & 31;
    const int lr    = lane & 15;
    const int kh    = (lane >> 4) << 4;
    const int rhalf = (lane >> 4) << 3;

    const int q0 = qt * 128 + w * 16;
    const float scale = 0.125f;          // 1/sqrt(64)
    const float NEG_INF = -__builtin_inff();

    // Q fragments (dq 0-31, 32-63)
    v16bf qf[2];
#pragma unroll
    for (int f = 0; f < 2; ++f)
        qf[f] = ldbf(Qp + ((size_t)bh * S_ + q0 + lr) * DH_ + f * 32 + kh);

    v8f acc[4];
#pragma unroll
    for (int i = 0; i < 4; ++i) acc[i] = vzero8();
    float mrun[8], lrun[8];
#pragma unroll
    for (int r = 0; r < 8; ++r) { mrun[r] = NEG_INF; lrun[r] = 0.f; }

    bf16* pl = &plds[w * 512];
    const bf16* kbase = Kp + (size_t)bh * S_ * DH_ + kh;
    const bf16* vbase = Vt + ((size_t)bh * DH_ + lr) * S_ + kh;
    const int kend = q0 + 16;            // keys needed: t <= q0+15 (causal)

#pragma unroll 1
    for (int t0 = 0; t0 < kend; t0 += 32) {
        const bf16* krow = kbase + (size_t)(t0 + lr) * DH_;
        // all 4 K fragments in one clause
        v16bf kf0 = ldbf(krow);
        v16bf kf1 = ldbf(krow + 32);
        v16bf kf2 = ldbf(krow + 16 * DH_);
        v16bf kf3 = ldbf(krow + 16 * DH_ + 32);
        // V fragments early: latency hidden behind the softmax VALU work
        v16bf vf[4];
#pragma unroll
        for (int nt = 0; nt < 4; ++nt)
            vf[nt] = ldbf(vbase + (size_t)nt * 16 * S_ + t0);
        if (t0 + 32 < kend) {
            __builtin_prefetch(krow + 32 * DH_, 0, 1);
            __builtin_prefetch(vbase + t0 + 32, 0, 1);
        }

        // scores: two 16x16 subtiles, DQ=64 reduced in two K=32 WMMAs each
        v8f s0 = vzero8(), s1 = vzero8();
        s0 = WMMA_BF16(qf[0], kf0, s0);
        s0 = WMMA_BF16(qf[1], kf1, s0);
        s1 = WMMA_BF16(qf[0], kf2, s1);
        s1 = WMMA_BF16(qf[1], kf3, s1);

        // causal mask + online softmax (row-wise over 16 lanes of each half)
#pragma unroll
        for (int r = 0; r < 8; ++r) {
            const int row = q0 + r + rhalf;
            const int c0  = t0 + lr;
            const int c1  = t0 + 16 + lr;
            float x0 = s0[r] * scale; if (c0 > row) x0 = NEG_INF;
            float x1 = s1[r] * scale; if (c1 > row) x1 = NEG_INF;
            float bm = fmaxf(x0, x1);
#pragma unroll
            for (int off = 1; off < 16; off <<= 1)
                bm = fmaxf(bm, __shfl_xor(bm, off, 32));
            const float mnew = fmaxf(mrun[r], bm);
            const float cfac = __expf(mrun[r] - mnew);
            const float p0 = __expf(x0 - mnew);
            const float p1 = __expf(x1 - mnew);
            float rs = p0 + p1;
#pragma unroll
            for (int off = 1; off < 16; off <<= 1)
                rs += __shfl_xor(rs, off, 32);
            lrun[r] = lrun[r] * cfac + rs;
            mrun[r] = mnew;
#pragma unroll
            for (int nt = 0; nt < 4; ++nt) acc[nt][r] *= cfac;
            const int prow = r + rhalf;
            pl[prow * 32 + lr]      = f2bf(p0);
            pl[prow * 32 + 16 + lr] = f2bf(p1);
        }

        // Re-load P in A-fragment layout (wave-private LDS; DScnt orders it).
        v16bf pf = ldbf(pl + lr * 32 + kh);

        // O += P[16x32] * V[32x64]  (V pre-transposed: contiguous columns)
#pragma unroll
        for (int nt = 0; nt < 4; ++nt)
            acc[nt] = WMMA_BF16(pf, vf[nt], acc[nt]);
    }

    // epilogue: normalize and store to Xo[b, s, h, dv]  (== [B*S, 1024] bf16)
    const int b = bh >> 4, h = bh & 15;
#pragma unroll
    for (int r = 0; r < 8; ++r) {
        const float inv = 1.0f / lrun[r];
        const int srow = q0 + r + rhalf;
#pragma unroll
        for (int nt = 0; nt < 4; ++nt) {
            const int dv = nt * 16 + lr;
            Xo[(((size_t)b * S_ + srow) * H_ + h) * DH_ + dv] = f2bf(acc[nt][r] * inv);
        }
    }
}

// ---------------------------------------------------------------------------
// Kernel 4: output projection  out = Xo * Wo^T + bo  (2-deep pipelined).
// grid = (B*S/128, D/64), block = 256.
// ---------------------------------------------------------------------------
__global__ __launch_bounds__(256) void outproj_kernel(
    const bf16* __restrict__ Xo, const bf16* __restrict__ wob,
    const float* __restrict__ bo, float* __restrict__ out) {
    const int rt = blockIdx.x;
    const int ct = blockIdx.y;
    const int w     = threadIdx.x >> 5;
    const int lane  = threadIdx.x & 31;
    const int lr    = lane & 15;
    const int kh    = (lane >> 4) << 4;
    const int rhalf = (lane >> 4) << 3;

    const int m0 = rt * 128 + w * 16;
    const int n0 = ct * 64;

    v8f acc[4];
#pragma unroll
    for (int i = 0; i < 4; ++i) acc[i] = vzero8();

    const bf16* arow = Xo + (size_t)(m0 + lr) * D_ + kh;
    const bf16* brow = wob + (size_t)(n0 + lr) * D_ + kh;   // + nt*16*D_ + kb

    auto ldA = [&](int kb, v16bf& af) { af = ldbf(arow + kb); };
    auto ldB = [&](int kb, v16bf bf[4]) {
#pragma unroll
        for (int nt = 0; nt < 4; ++nt)
            bf[nt] = ldbf(brow + (size_t)nt * 16 * D_ + kb);
    };

    v16bf af0, af1, b0[4], b1[4];
    ldA(0, af0);
    ldB(0, b0);
#pragma unroll 1
    for (int kb = 0; kb < D_; kb += 64) {
        ldA(kb + 32, af1);
        ldB(kb + 32, b1);
        if (kb + 96 < D_) __builtin_prefetch(arow + kb + 96, 0, 1);
#pragma unroll
        for (int nt = 0; nt < 4; ++nt) acc[nt] = WMMA_BF16(af0, b0[nt], acc[nt]);
        if (kb + 64 < D_) {
            ldA(kb + 64, af0);
            ldB(kb + 64, b0);
        }
#pragma unroll
        for (int nt = 0; nt < 4; ++nt) acc[nt] = WMMA_BF16(af1, b1[nt], acc[nt]);
    }

#pragma unroll
    for (int nt = 0; nt < 4; ++nt) {
        const int col = n0 + nt * 16 + lr;
        const float bias = bo[col];
#pragma unroll
        for (int r = 0; r < 8; ++r) {
            const int row = m0 + r + rhalf;
            out[(size_t)row * D_ + col] = acc[nt][r] + bias;
        }
    }
}

// ---------------------------------------------------------------------------
// Host launch.  Inputs (setup_inputs order): q,k,v,Wq,Wk,Wv,Wo,bo (all f32).
// Workspace (bf16): Wq_t | Wk_t | Wv_t | Wo | Qp | Kp | Vt | Xo  (~75.5 MB).
// ---------------------------------------------------------------------------
extern "C" void kernel_launch(void* const* d_in, const int* in_sizes, int n_in,
                              void* d_out, int out_size, void* d_ws, size_t ws_size,
                              hipStream_t stream) {
    (void)in_sizes; (void)n_in; (void)out_size; (void)ws_size;

    const float* q  = (const float*)d_in[0];
    const float* k  = (const float*)d_in[1];
    const float* v  = (const float*)d_in[2];
    const float* Wq = (const float*)d_in[3];
    const float* Wk = (const float*)d_in[4];
    const float* Wv = (const float*)d_in[5];
    const float* Wo = (const float*)d_in[6];
    const float* bo = (const float*)d_in[7];
    float* out = (float*)d_out;

    const size_t WELEM = (size_t)H_ * D_ * DH_;        // 1,048,576
    const size_t PELEM = (size_t)B_ * H_ * S_ * DH_;   // 8,388,608

    bf16* wqt = (bf16*)d_ws;
    bf16* wkt = wqt + WELEM;
    bf16* wvt = wkt + WELEM;
    bf16* wob = wvt + WELEM;
    bf16* Qp  = wob + (size_t)D_ * D_;
    bf16* Kp  = Qp + PELEM;
    bf16* Vt  = Kp + PELEM;
    bf16* Xo  = Vt + PELEM;

    cvt_weights_kernel<<<2048, 256, 0, stream>>>(Wq, Wk, Wv, Wo, wqt, wkt, wvt, wob);
    proj_kernel<<<dim3(S_ / 128, B_ * H_, 3), 256, 0, stream>>>(
        q, k, v, wqt, wkt, wvt, Qp, Kp, Vt);
    attn_kernel<<<dim3(S_ / 128, B_ * H_), 256, 0, stream>>>(Qp, Kp, Vt, Xo);
    outproj_kernel<<<dim3((B_ * S_) / 128, D_ / 64), 256, 0, stream>>>(
        Xo, wob, bo, out);
}
